// StrataTier_20160576487585
// MI455X (gfx1250) — matrix operand order repaired and checked by reading, not compiled, not run on hardware
//
#include <hip/hip_runtime.h>
#include <hip/hip_bf16.h>
#include <math.h>

// Problem constants (match reference)
#define N_TOK   128
#define BATCH   2
#define DIN_    128
#define DMODEL  256
#define DFF_    1024
#define NHEAD_  8
#define DH_     32
#define NLAYERS_ 2
#define BLOCKW  32
#define WMAX    65          // 2*BLOCK+1
#define LEN_    160
#define ROWS_   (N_TOK*WMAX*BATCH)   // 16640 rows; 520 tiles of 32

typedef __attribute__((ext_vector_type(16))) _Float16 v16h;
typedef __attribute__((ext_vector_type(8)))  _Float16 h8;
typedef __attribute__((ext_vector_type(8)))  float    v8f;

// ---------------------------------------------------------------------------
// Convert f32 weight (K x N, row major) -> f16 transposed (N x K, row major)
// so GEMM B-fragments become contiguous 16B loads.
__global__ void k_w_to_f16t(const float* __restrict__ W, _Float16* __restrict__ Wt,
                            int K, int N) {
    int id = blockIdx.x * 256 + threadIdx.x;
    if (id >= K * N) return;
    int n = id % N;
    int k = id / N;
    Wt[(size_t)n * K + k] = (_Float16)W[id];
}

// ---------------------------------------------------------------------------
// x[n][w][b][d] = inputs[n][b][d] + PE(pos=cols[n][w], d), stored f16
__global__ void k_build_x(const float* __restrict__ inp, _Float16* __restrict__ x) {
    int id = blockIdx.x * 256 + threadIdx.x;
    if (id >= ROWS_ * DIN_) return;
    int d  = id % DIN_;
    int r  = id / DIN_;
    int b  = r & 1;
    int nw = r >> 1;
    int w  = nw % WMAX;
    int n  = nw / WMAX;
    int s  = n - BLOCKW; if (s < 0) s = 0;
    float pos  = (float)(s + w);
    float freq = expf((float)(d & ~1) * (-9.210340371976184f / (float)DIN_)); // ln(1e4)
    float ang  = pos * freq;
    float pe   = (d & 1) ? cosf(ang) : sinf(ang);
    x[id] = (_Float16)(inp[(size_t)(n * BATCH + b) * DIN_ + d] + pe);
}

// ---------------------------------------------------------------------------
// Fragment loaders (ISA 16-bit WMMA operand layouts, wave32)
// A: lane holds row M=lane&15; element e -> K = (e/8)*16 + (lane>>4)*8 + e%8
__device__ __forceinline__ v16h load_afrag(const _Float16* __restrict__ rowp,
                                           int k0, int hf) {
    h8 lo = *(const h8*)(rowp + k0 + hf * 8);
    h8 hi = *(const h8*)(rowp + k0 + 16 + hf * 8);
    v16h a;
    #pragma unroll
    for (int i = 0; i < 8; i++) { a[i] = lo[i]; a[i + 8] = hi[i]; }
    return a;
}
// B (from N x K transposed weights): lane holds col N=lane&15; elem e -> K = (e/8)*16 + (lane>>4)*8 + e%8
__device__ __forceinline__ v16h load_bfrag(const _Float16* __restrict__ colp,
                                           int k0, int hf) {
    h8 b0 = *(const h8*)(colp + k0 + hf * 8);
    h8 b1 = *(const h8*)(colp + k0 + 16 + hf * 8);
    v16h b;
    #pragma unroll
    for (int i = 0; i < 8; i++) { b[i] = b0[i]; b[i + 8] = b1[i]; }
    return b;
}

// ---------------------------------------------------------------------------
// Wave-level WMMA GEMM: C[M x N] = act(A[M x K] * Bt[N x K]^T + bias)
// One wave per 32(M) x 64(N) strip. B fragments loaded one at a time (short
// live ranges -> no spills); 8 WMMAs per 32-wide k step.
// K % 32 == 0, N % 64 == 0, M % 32 == 0. Optional secondary f16 output C2.
template <bool RELU, typename OT>
__global__ void __launch_bounds__(32)
k_gemm(const _Float16* __restrict__ A, const _Float16* __restrict__ Bt,
       const float* __restrict__ bias, OT* __restrict__ C,
       _Float16* __restrict__ C2, int M, int Nn, int K) {
    const int lane = threadIdx.x;        // wave32
    const int lm   = lane & 15;
    const int hf   = lane >> 4;
    const int mB   = blockIdx.x * 32;
    const int nB   = blockIdx.y * 64;

    const _Float16* a0p = A + (size_t)(mB + lm) * K;
    const _Float16* a1p = A + (size_t)(mB + 16 + lm) * K;
    const _Float16* bp0 = Bt + (size_t)(nB + 0 * 16 + lm) * K;
    const _Float16* bp1 = Bt + (size_t)(nB + 1 * 16 + lm) * K;
    const _Float16* bp2 = Bt + (size_t)(nB + 2 * 16 + lm) * K;
    const _Float16* bp3 = Bt + (size_t)(nB + 3 * 16 + lm) * K;

    v8f acc[2][4] = {};

    #pragma unroll 1
    for (int k0 = 0; k0 < K; k0 += 32) {
        __builtin_prefetch(a0p + k0 + 128, 0, 1);   // -> global_prefetch_b8
        __builtin_prefetch(a1p + k0 + 128, 0, 1);
        v16h a0 = load_afrag(a0p, k0, hf);
        v16h a1 = load_afrag(a1p, k0, hf);
        v16h bf;
        bf = load_bfrag(bp0, k0, hf);
        acc[0][0] = __builtin_amdgcn_wmma_f32_16x16x32_f16(false, a0, false, bf, (short)0, acc[0][0], false, false);
        acc[1][0] = __builtin_amdgcn_wmma_f32_16x16x32_f16(false, a1, false, bf, (short)0, acc[1][0], false, false);
        bf = load_bfrag(bp1, k0, hf);
        acc[0][1] = __builtin_amdgcn_wmma_f32_16x16x32_f16(false, a0, false, bf, (short)0, acc[0][1], false, false);
        acc[1][1] = __builtin_amdgcn_wmma_f32_16x16x32_f16(false, a1, false, bf, (short)0, acc[1][1], false, false);
        bf = load_bfrag(bp2, k0, hf);
        acc[0][2] = __builtin_amdgcn_wmma_f32_16x16x32_f16(false, a0, false, bf, (short)0, acc[0][2], false, false);
        acc[1][2] = __builtin_amdgcn_wmma_f32_16x16x32_f16(false, a1, false, bf, (short)0, acc[1][2], false, false);
        bf = load_bfrag(bp3, k0, hf);
        acc[0][3] = __builtin_amdgcn_wmma_f32_16x16x32_f16(false, a0, false, bf, (short)0, acc[0][3], false, false);
        acc[1][3] = __builtin_amdgcn_wmma_f32_16x16x32_f16(false, a1, false, bf, (short)0, acc[1][3], false, false);
    }

    #pragma unroll
    for (int mi = 0; mi < 2; mi++) {
        #pragma unroll
        for (int j = 0; j < 4; j++) {
            int col  = nB + j * 16 + lm;
            float bv = bias ? bias[col] : 0.0f;
            #pragma unroll
            for (int v = 0; v < 8; v++) {
                int row = mB + mi * 16 + hf * 8 + v;   // C/D: M = (lane/16)*8 + vgpr
                float val = acc[mi][j][v] + bv;
                if (RELU) val = fmaxf(val, 0.0f);
                C[(size_t)row * Nn + col] = (OT)val;
                if (C2) C2[(size_t)row * Nn + col] = (_Float16)val;
            }
        }
    }
}

// ---------------------------------------------------------------------------
// Windowed attention: one block per (n, b, head). Flash-style online softmax.
// qkv is f16 (f32 accumulate); output written f16 (feeds Wo GEMM only).
__global__ void k_attn(const _Float16* __restrict__ qkv, _Float16* __restrict__ o) {
    int blk = blockIdx.x;
    int hh  = blk % NHEAD_;
    int nb  = blk / NHEAD_;
    int b   = nb % BATCH;
    int n   = nb / BATCH;

    __shared__ float sk[WMAX][DH_];
    __shared__ float sv[WMAX][DH_];
    int t = threadIdx.x;
    for (int idx = t; idx < WMAX * DH_; idx += blockDim.x) {
        int w = idx / DH_, e = idx % DH_;
        size_t r = ((size_t)(n * WMAX + w) * BATCH + b) * (3 * DMODEL);
        sk[w][e] = (float)qkv[r + DMODEL     + hh * DH_ + e];
        sv[w][e] = (float)qkv[r + 2 * DMODEL + hh * DH_ + e];
    }
    __syncthreads();

    int s    = n - BLOCKW; if (s < 0) s = 0;
    int endv = n + BLOCKW + 1; if (endv > LEN_) endv = LEN_;
    int nvalid = endv - s;                 // key ki valid iff ki < nvalid

    if (t < WMAX) {
        int qi = t;
        size_t r = ((size_t)(n * WMAX + qi) * BATCH + b) * (3 * DMODEL);
        float q[DH_];
        #pragma unroll
        for (int e = 0; e < DH_; e++) q[e] = (float)qkv[r + hh * DH_ + e];

        const float scale = 0.17677669529663687f;  // 1/sqrt(32)
        float m = -INFINITY, den = 0.0f, acc[DH_];
        #pragma unroll
        for (int e = 0; e < DH_; e++) acc[e] = 0.0f;

        for (int ki = 0; ki < nvalid; ki++) {
            float x = 0.0f;
            #pragma unroll
            for (int e = 0; e < DH_; e++) x += q[e] * sk[ki][e];
            x *= scale;
            float mn = fmaxf(m, x);
            float f  = expf(m - mn);   // expf(-inf)=0 on first hit
            float p  = expf(x - mn);
            den = den * f + p;
            #pragma unroll
            for (int e = 0; e < DH_; e++) acc[e] = acc[e] * f + p * sv[ki][e];
            m = mn;
        }
        float inv = 1.0f / den;
        size_t orow = ((size_t)(n * WMAX + qi) * BATCH + b) * DMODEL;
        #pragma unroll
        for (int e = 0; e < DH_; e++) o[orow + hh * DH_ + e] = (_Float16)(acc[e] * inv);
    }
}

// ---------------------------------------------------------------------------
// h = LayerNorm(h + add) * g + beta; also emits f16 copy for downstream GEMM A.
__global__ void k_ln_res(float* __restrict__ h, const float* __restrict__ add,
                         const float* __restrict__ g, const float* __restrict__ beta,
                         _Float16* __restrict__ h16) {
    __shared__ float red[DMODEL];
    int r = blockIdx.x, t = threadIdx.x;
    float y = h[(size_t)r * DMODEL + t] + add[(size_t)r * DMODEL + t];
    red[t] = y; __syncthreads();
    for (int sft = DMODEL / 2; sft > 0; sft >>= 1) {
        if (t < sft) red[t] += red[t + sft];
        __syncthreads();
    }
    float mean = red[0] / (float)DMODEL;
    __syncthreads();
    float d = y - mean;
    red[t] = d * d; __syncthreads();
    for (int sft = DMODEL / 2; sft > 0; sft >>= 1) {
        if (t < sft) red[t] += red[t + sft];
        __syncthreads();
    }
    float var = red[0] / (float)DMODEL;
    float outv = d * rsqrtf(var + 1e-5f) * g[t] + beta[t];
    h[(size_t)r * DMODEL + t]   = outv;
    h16[(size_t)r * DMODEL + t] = (_Float16)outv;
}

// ---------------------------------------------------------------------------
// kp = relu(inputs @ k_W1 + k_b1) @ k_W2 + k_b2  -> (N_TOK*BATCH, 2)
__global__ void k_kp(const float* __restrict__ inp, const float* __restrict__ W1,
                     const float* __restrict__ b1, const float* __restrict__ W2,
                     const float* __restrict__ b2, float* __restrict__ kp) {
    __shared__ float si[DIN_];
    __shared__ float r0[256], r1[256];
    int r = blockIdx.x;     // n*BATCH + b
    int t = threadIdx.x;
    if (t < DIN_) si[t] = inp[(size_t)r * DIN_ + t];
    __syncthreads();
    float p0 = 0.0f, p1 = 0.0f;
    for (int j = t; j < DFF_; j += 256) {
        float hs = b1[j];
        for (int d = 0; d < DIN_; d++) hs += si[d] * W1[(size_t)d * DFF_ + j];
        hs = fmaxf(hs, 0.0f);
        p0 += hs * W2[(size_t)j * 2 + 0];
        p1 += hs * W2[(size_t)j * 2 + 1];
    }
    r0[t] = p0; r1[t] = p1; __syncthreads();
    for (int sft = 128; sft > 0; sft >>= 1) {
        if (t < sft) { r0[t] += r0[t + sft]; r1[t] += r1[t + sft]; }
        __syncthreads();
    }
    if (t == 0) { kp[r * 2 + 0] = r0[0] + b2[0]; kp[r * 2 + 1] = r1[0] + b2[1]; }
}

// ---------------------------------------------------------------------------
// Final: out[col,b,:] = sum_n softmax_n(hier) * h[n, w(n,col), b, :]
// Inverted scatter: n participates iff w = col - start(n) in [0,65) and col < n+33.
__global__ void k_final(const float* __restrict__ h, const float* __restrict__ kp,
                        float* __restrict__ out) {
    int col = blockIdx.x, b = blockIdx.y, d = threadIdx.x;
    float m = -INFINITY;
    for (int n = 0; n < N_TOK; n++) {
        int s = n - BLOCKW; if (s < 0) s = 0;
        int w = col - s;
        if (w < 0 || w >= WMAX || col >= n + BLOCKW + 1) continue;
        float kp0 = kp[(n * BATCH + b) * 2 + 0];
        float kp1 = kp[(n * BATCH + b) * 2 + 1];
        float diff = (float)col - 1.25f * (float)n - kp0;   // hscale = 160/128
        float logit = -diff * diff * expf(kp1);
        m = fmaxf(m, logit);
    }
    float den = 0.0f, num = 0.0f;
    for (int n = 0; n < N_TOK; n++) {
        int s = n - BLOCKW; if (s < 0) s = 0;
        int w = col - s;
        if (w < 0 || w >= WMAX || col >= n + BLOCKW + 1) continue;
        float kp0 = kp[(n * BATCH + b) * 2 + 0];
        float kp1 = kp[(n * BATCH + b) * 2 + 1];
        float diff = (float)col - 1.25f * (float)n - kp0;
        float logit = -diff * diff * expf(kp1);
        float p = expf(logit - m);
        den += p;
        num += p * h[((size_t)(n * WMAX + w) * BATCH + b) * DMODEL + d];
    }
    out[((size_t)col * BATCH + b) * DMODEL + d] = num / den;
}

// ---------------------------------------------------------------------------
extern "C" void kernel_launch(void* const* d_in, const int* in_sizes, int n_in,
                              void* d_out, int out_size, void* d_ws, size_t ws_size,
                              hipStream_t stream) {
    (void)in_sizes; (void)n_in; (void)out_size; (void)ws_size;
    const float* inputs = (const float*)d_in[0];
    const float* in_W1  = (const float*)d_in[1];
    const float* in_b1  = (const float*)d_in[2];
    const float* in_W2  = (const float*)d_in[3];
    const float* in_b2  = (const float*)d_in[4];
    const float* k_W1   = (const float*)d_in[5];
    const float* k_b1   = (const float*)d_in[6];
    const float* k_W2   = (const float*)d_in[7];
    const float* k_b2   = (const float*)d_in[8];
    const float* Wqkv   = (const float*)d_in[9];
    const float* bqkv   = (const float*)d_in[10];
    const float* Wo     = (const float*)d_in[11];
    const float* bo     = (const float*)d_in[12];
    const float* ln1_g  = (const float*)d_in[13];
    const float* ln1_b  = (const float*)d_in[14];
    const float* W1     = (const float*)d_in[15];
    const float* b1     = (const float*)d_in[16];
    const float* W2     = (const float*)d_in[17];
    const float* b2     = (const float*)d_in[18];
    const float* ln2_g  = (const float*)d_in[19];
    const float* ln2_b  = (const float*)d_in[20];
    float* out = (float*)d_out;

    // ---- carve workspace ----
    char* ws = (char*)d_ws;
    size_t off = 0;
    auto carve = [&](size_t bytes) -> void* {
        void* p = ws + off;
        off += (bytes + 255) & ~(size_t)255;
        return p;
    };
    _Float16* x16    = (_Float16*)carve((size_t)ROWS_ * DIN_ * 2);
    _Float16* hid16  = (_Float16*)carve((size_t)ROWS_ * DFF_ * 2);
    float*    hbuf   = (float*)   carve((size_t)ROWS_ * DMODEL * 4);
    _Float16* h16    = (_Float16*)carve((size_t)ROWS_ * DMODEL * 2);
    _Float16* qkv16  = (_Float16*)carve((size_t)ROWS_ * 3 * DMODEL * 2);
    _Float16* o16    = (_Float16*)carve((size_t)ROWS_ * DMODEL * 2);
    float*    tmp256 = (float*)   carve((size_t)ROWS_ * DMODEL * 4);
    float*    kpb    = (float*)   carve((size_t)N_TOK * BATCH * 2 * 4);
    _Float16* inW1t  = (_Float16*)carve((size_t)DFF_ * DIN_ * 2);
    _Float16* inW2t  = (_Float16*)carve((size_t)DMODEL * DFF_ * 2);
    _Float16* qkvWt  = (_Float16*)carve((size_t)NLAYERS_ * 3 * DMODEL * DMODEL * 2);
    _Float16* WoT    = (_Float16*)carve((size_t)NLAYERS_ * DMODEL * DMODEL * 2);
    _Float16* W1t    = (_Float16*)carve((size_t)NLAYERS_ * DFF_ * DMODEL * 2);
    _Float16* W2t    = (_Float16*)carve((size_t)NLAYERS_ * DMODEL * DFF_ * 2);

    auto conv = [&](const float* W, _Float16* Wt, int K, int N) {
        int tot = K * N;
        k_w_to_f16t<<<(tot + 255) / 256, 256, 0, stream>>>(W, Wt, K, N);
    };

    // ---- weight conversion (transposed f16) ----
    conv(in_W1, inW1t, DIN_, DFF_);
    conv(in_W2, inW2t, DFF_, DMODEL);
    for (int l = 0; l < NLAYERS_; l++) {
        conv(Wqkv + (size_t)l * DMODEL * 3 * DMODEL, qkvWt + (size_t)l * 3 * DMODEL * DMODEL, DMODEL, 3 * DMODEL);
        conv(Wo   + (size_t)l * DMODEL * DMODEL,     WoT   + (size_t)l * DMODEL * DMODEL,     DMODEL, DMODEL);
        conv(W1   + (size_t)l * DMODEL * DFF_,       W1t   + (size_t)l * DFF_ * DMODEL,       DMODEL, DFF_);
        conv(W2   + (size_t)l * DFF_ * DMODEL,       W2t   + (size_t)l * DMODEL * DFF_,       DFF_, DMODEL);
    }

    // ---- build x = inputs + PE (f16) ----
    k_build_x<<<(ROWS_ * DIN_ + 255) / 256, 256, 0, stream>>>(inputs, x16);

    const int MT = ROWS_ / 32;   // 520
    dim3 blk(32);

    // ---- input MLP ----
    k_gemm<true, _Float16><<<dim3(MT, DFF_ / 64), blk, 0, stream>>>(
        x16, inW1t, in_b1, hid16, (_Float16*)nullptr, ROWS_, DFF_, DIN_);
    k_gemm<false, float><<<dim3(MT, DMODEL / 64), blk, 0, stream>>>(
        hid16, inW2t, in_b2, hbuf, h16, ROWS_, DMODEL, DFF_);

    // ---- transformer layers ----
    for (int l = 0; l < NLAYERS_; l++) {
        k_gemm<false, _Float16><<<dim3(MT, (3 * DMODEL) / 64), blk, 0, stream>>>(
            h16, qkvWt + (size_t)l * 3 * DMODEL * DMODEL, bqkv + (size_t)l * 3 * DMODEL,
            qkv16, (_Float16*)nullptr, ROWS_, 3 * DMODEL, DMODEL);
        k_attn<<<N_TOK * BATCH * NHEAD_, 128, 0, stream>>>(qkv16, o16);
        k_gemm<false, float><<<dim3(MT, DMODEL / 64), blk, 0, stream>>>(
            o16, WoT + (size_t)l * DMODEL * DMODEL, bo + (size_t)l * DMODEL,
            tmp256, (_Float16*)nullptr, ROWS_, DMODEL, DMODEL);
        k_ln_res<<<ROWS_, DMODEL, 0, stream>>>(hbuf, tmp256,
            ln1_g + (size_t)l * DMODEL, ln1_b + (size_t)l * DMODEL, h16);
        k_gemm<true, _Float16><<<dim3(MT, DFF_ / 64), blk, 0, stream>>>(
            h16, W1t + (size_t)l * DFF_ * DMODEL, b1 + (size_t)l * DFF_,
            hid16, (_Float16*)nullptr, ROWS_, DFF_, DMODEL);
        k_gemm<false, float><<<dim3(MT, DMODEL / 64), blk, 0, stream>>>(
            hid16, W2t + (size_t)l * DMODEL * DFF_, b2 + (size_t)l * DMODEL,
            tmp256, (_Float16*)nullptr, ROWS_, DMODEL, DFF_);
        k_ln_res<<<ROWS_, DMODEL, 0, stream>>>(hbuf, tmp256,
            ln2_g + (size_t)l * DMODEL, ln2_b + (size_t)l * DMODEL, h16);
    }

    // ---- kernel-position MLP ----
    k_kp<<<N_TOK * BATCH, 256, 0, stream>>>(inputs, k_W1, k_b1, k_W2, k_b2, kpb);

    // ---- final scatter/softmax gather ----
    k_final<<<dim3(LEN_, BATCH), DMODEL, 0, stream>>>(hbuf, kpb, out);
}
